// ConvNet_87428354277500
// MI455X (gfx1250) — matrix operand order repaired and checked by reading, not compiled, not run on hardware
//
#include <hip/hip_runtime.h>

typedef __attribute__((ext_vector_type(16))) _Float16 v16h;
typedef __attribute__((ext_vector_type(8)))  float    v8f;

#define B_TOTAL 131072

// fragment counts: fc1 K=384 (12 kt) x N=256 (16 nt); fc2 K=256 x N=64; head K=64 x N=16
#define F1_FRAGS (12*16)
#define F2_FRAGS (8*4)
#define HD_FRAGS (2*1)
#define F1_HALF (F1_FRAGS*32*16)
#define F2_HALF (F2_FRAGS*32*16)
#define HD_HALF (HD_FRAGS*32*16)

// per-wave LDS layout (bytes)
#define WAVE_LDS 34816
#define HDR_LDS  1024
#define OFF_R1   0        // xs (16x150 f32 = 9600) then h1 (16x190 f32 = 12160)
#define OFF_R2   12288    // sd (16x120 f32 = 7680) then Aflat frags (12*32*16 f16 = 12288) then logits
#define OFF_AFC2 24576    // 8*32*16 f16 = 8192
#define OFF_AHEAD 32768   // 2*32*16 f16 = 2048

// ---------------- weight prep: f32 -> f16 B-fragments in d_ws ----------------
// B-matrix fragment layout (16x16x32 f16 WMMA): col = lane&15, K = (e&7) + ((e>>3)<<4) + (lane>=16)*8
__global__ void prep_weights(const float* __restrict__ fc1_w,
                             const float* __restrict__ fc2_w,
                             const float* __restrict__ pamap_w,
                             _Float16* __restrict__ ws)
{
    int idx = blockIdx.x * blockDim.x + threadIdx.x;
    const int total = F1_HALF + F2_HALF + HD_HALF;
    if (idx >= total) return;
    float val;
    if (idx < F1_HALF) {
        int e = idx & 15, ln = (idx >> 4) & 31, f = idx >> 9;
        int nt = f & 15, kt = f >> 4;
        int k = kt*32 + (e & 7) + ((e >> 3) << 4) + ((ln >> 4) << 3);
        int n = nt*16 + (ln & 15);
        val = (k < 360) ? fc1_w[n*360 + k] : 0.f;
    } else if (idx < F1_HALF + F2_HALF) {
        int r = idx - F1_HALF;
        int e = r & 15, ln = (r >> 4) & 31, f = r >> 9;
        int nt = f & 3, kt = f >> 2;
        int k = kt*32 + (e & 7) + ((e >> 3) << 4) + ((ln >> 4) << 3);
        int n = nt*16 + (ln & 15);
        val = fc2_w[n*256 + k];
    } else {
        int r = idx - F1_HALF - F2_HALF;
        int e = r & 15, ln = (r >> 4) & 31, kt = r >> 9;
        int k = kt*32 + (e & 7) + ((e >> 3) << 4) + ((ln >> 4) << 3);
        int n = ln & 15;
        val = (n < 12) ? pamap_w[n*64 + k] : 0.f;
    }
    ws[idx] = (_Float16)val;
}

// ---------------- fused pipeline: 8 waves/block, 16 samples/wave ----------------
__launch_bounds__(256)
__global__ void convnet_main(const float* __restrict__ sig,
                             const float* __restrict__ c1w, const float* __restrict__ c1b,
                             const float* __restrict__ c2w, const float* __restrict__ c2b,
                             const float* __restrict__ fc1_b,
                             const float* __restrict__ fc2_b,
                             const float* __restrict__ pamap_b,
                             const _Float16* __restrict__ ws,
                             float* __restrict__ out)
{
    extern __shared__ __attribute__((aligned(32))) char smem[];
    const int tid  = threadIdx.x;
    const int wave = tid >> 5;
    const int lane = tid & 31;

    // block-shared conv weights: [0,45) w1, [45,50) b1, [50,200) w2, [200,210) b2
    float* wconv = (float*)smem;
    for (int i = tid; i < 210; i += 256) {
        float v;
        if      (i < 45)  v = c1w[i];
        else if (i < 50)  v = c1b[i - 45];
        else if (i < 200) v = c2w[i - 50];
        else              v = c2b[i - 200];
        wconv[i] = v;
    }

    char* wbase = smem + HDR_LDS + wave * WAVE_LDS;
    float*    xs     = (float*)(wbase + OFF_R1);
    float*    h1     = (float*)(wbase + OFF_R1);
    float*    sd     = (float*)(wbase + OFF_R2);
    _Float16* Aflat  = (_Float16*)(wbase + OFF_R2);
    float*    logits = (float*)(wbase + OFF_R2);
    _Float16* Afc2   = (_Float16*)(wbase + OFF_AFC2);
    _Float16* Ahead  = (_Float16*)(wbase + OFF_AHEAD);

    const int tile = blockIdx.x * 8 + wave;     // 16-sample tile index
    // stage 0: coalesced signal load (16 samples * 150 f32 = 2400 f32, 16B aligned per tile)
    {
        const float4* src = (const float4*)(sig + (long long)tile * 2400);
        float4* dst = (float4*)xs;
        for (int i = lane; i < 600; i += 32) dst[i] = src[i];
    }
    __syncthreads();

    const int s  = lane >> 1;   // sample within tile
    const int hh = lane & 1;    // which half of the time axis

    // stage 1: running std (window 10, ddof=1), sliding sums
    {
        const float* x   = xs + s * 150;
        float*       sdp = sd + s * 120;
        const int t0 = hh * 20;
        for (int c = 0; c < 3; ++c) {
            float sum = 0.f, ss = 0.f;
#pragma unroll
            for (int w = 0; w < 10; ++w) { float v = x[(t0 + w)*3 + c]; sum += v; ss += v*v; }
            for (int t = t0; t < t0 + 20; ++t) {
                float var = (ss - sum*sum*0.1f) * (1.f/9.f);
                sdp[t*3 + c] = sqrtf(fmaxf(var, 0.f));
                float vo = x[t*3 + c], vi = x[(t + 10)*3 + c];
                sum += vi - vo; ss += vi*vi - vo*vo;
            }
        }
    }
    __syncthreads();

    // stage 2: conv1 (5 out, k=3) + relu   sd(40x3) -> h1(5x38)
    {
        const float* sdp = sd + s * 120;
        float*       hp  = h1 + s * 190;
        const int t0 = hh * 19;
        for (int o = 0; o < 5; ++o) {
            float w[9];
#pragma unroll
            for (int j = 0; j < 9; ++j) w[j] = wconv[o*9 + j];
            float b = wconv[45 + o];
            for (int t = t0; t < t0 + 19; ++t) {
                float acc = b;
#pragma unroll
                for (int c = 0; c < 3; ++c)
#pragma unroll
                    for (int k = 0; k < 3; ++k)
                        acc += w[c*3 + k] * sdp[(t + k)*3 + c];
                hp[o*38 + t] = fmaxf(acc, 0.f);
            }
        }
    }
    __syncthreads();

    // stage 3: conv2 (10 out) + relu -> flat[k=t*10+o], written straight into
    // WMMA A-fragment layout: row = lane&15, K = (e&7) + ((e>>3)<<4) + ((lane>>4)<<3)
    {
        const float* hp = h1 + s * 190;
        const int t0 = hh * 18;
        for (int o = 0; o < 10; ++o) {
            float w[15];
#pragma unroll
            for (int j = 0; j < 15; ++j) w[j] = wconv[50 + o*15 + j];
            float b = wconv[200 + o];
            for (int t = t0; t < t0 + 18; ++t) {
                float acc = b;
#pragma unroll
                for (int i = 0; i < 5; ++i)
#pragma unroll
                    for (int k = 0; k < 3; ++k)
                        acc += w[i*3 + k] * hp[i*38 + t + k];
                acc = fmaxf(acc, 0.f);
                int kidx = t*10 + o;
                int kt = kidx >> 5, r = kidx & 31;
                int e  = (r & 7) | ((r >> 1) & 8);
                int ln = (((r >> 3) & 1) << 4) | s;
                Aflat[(kt*32 + ln)*16 + e] = (_Float16)acc;
            }
        }
        // zero-pad K 360..383 for all 16 rows
        for (int j = lane; j < 384; j += 32) {
            int kidx = 360 + (j >> 4);
            int s2 = j & 15;
            int kt = kidx >> 5, r = kidx & 31;
            int e  = (r & 7) | ((r >> 1) & 8);
            int ln = (((r >> 3) & 1) << 4) | s2;
            Aflat[(kt*32 + ln)*16 + e] = (_Float16)0.f;
        }
    }
    __syncthreads();

    const int col    = lane & 15;
    const int rowoff = (lane >> 4) << 3;   // C/D fragment: elem v -> row = rowoff + v

    // stage 4: fc1  (16x256x384) via 16 N-tiles x 12 chained WMMAs
    {
        v16h a[12];
#pragma unroll
        for (int kt = 0; kt < 12; ++kt)
            a[kt] = *(const v16h*)(Aflat + (kt*32 + lane)*16);
        const _Float16* wsF1 = ws;
#pragma unroll 1
        for (int nt = 0; nt < 16; ++nt) {
            float bias = fc1_b[nt*16 + col];
            v8f acc;
#pragma unroll
            for (int v = 0; v < 8; ++v) acc[v] = bias;
#pragma unroll
            for (int kt = 0; kt < 12; ++kt) {
                v16h bf = *(const v16h*)(wsF1 + ((kt*16 + nt)*32 + lane)*16);
                acc = __builtin_amdgcn_wmma_f32_16x16x32_f16(false, a[kt], false, bf,
                                                             (short)0, acc, false, false);
            }
#pragma unroll
            for (int v = 0; v < 8; ++v) {
                float r = fmaxf(acc[v], 0.f);
                int srow = rowoff + v;
                int k2 = nt*16 + col;
                int kt2 = k2 >> 5, r2 = k2 & 31;
                int e2  = (r2 & 7) | ((r2 >> 1) & 8);
                int ln2 = (((r2 >> 3) & 1) << 4) | srow;
                Afc2[(kt2*32 + ln2)*16 + e2] = (_Float16)r;
            }
        }
    }
    __syncthreads();

    // stage 5: fc2 (16x64x256): 4 N-tiles x 8 WMMAs
    {
        v16h a[8];
#pragma unroll
        for (int kt = 0; kt < 8; ++kt)
            a[kt] = *(const v16h*)(Afc2 + (kt*32 + lane)*16);
        const _Float16* wsF2 = ws + F1_HALF;
#pragma unroll 1
        for (int nt = 0; nt < 4; ++nt) {
            float bias = fc2_b[nt*16 + col];
            v8f acc;
#pragma unroll
            for (int v = 0; v < 8; ++v) acc[v] = bias;
#pragma unroll
            for (int kt = 0; kt < 8; ++kt) {
                v16h bf = *(const v16h*)(wsF2 + ((kt*4 + nt)*32 + lane)*16);
                acc = __builtin_amdgcn_wmma_f32_16x16x32_f16(false, a[kt], false, bf,
                                                             (short)0, acc, false, false);
            }
#pragma unroll
            for (int v = 0; v < 8; ++v) {
                float r = fmaxf(acc[v], 0.f);
                int srow = rowoff + v;
                int k3 = nt*16 + col;
                int kt3 = k3 >> 5, r3 = k3 & 31;
                int e3  = (r3 & 7) | ((r3 >> 1) & 8);
                int ln3 = (((r3 >> 3) & 1) << 4) | srow;
                Ahead[(kt3*32 + ln3)*16 + e3] = (_Float16)r;
            }
        }
    }
    __syncthreads();

    // stage 6: pamap head (16x16x64, N padded; padded cols get 0 weights/bias)
    {
        v16h a0 = *(const v16h*)(Ahead + (0*32 + lane)*16);
        v16h a1 = *(const v16h*)(Ahead + (1*32 + lane)*16);
        const _Float16* wsHd = ws + F1_HALF + F2_HALF;
        float bias = (col < 12) ? pamap_b[col] : 0.f;
        v8f acc;
#pragma unroll
        for (int v = 0; v < 8; ++v) acc[v] = bias;
        v16h b0 = *(const v16h*)(wsHd + (0*32 + lane)*16);
        v16h b1 = *(const v16h*)(wsHd + (1*32 + lane)*16);
        acc = __builtin_amdgcn_wmma_f32_16x16x32_f16(false, a0, false, b0, (short)0, acc, false, false);
        acc = __builtin_amdgcn_wmma_f32_16x16x32_f16(false, a1, false, b1, (short)0, acc, false, false);
#pragma unroll
        for (int v = 0; v < 8; ++v)
            logits[(rowoff + v)*16 + col] = acc[v];
    }
    __syncthreads();

    // stage 7: log_softmax over 12 classes, one sample per lane (lanes 0..15)
    if (lane < 16) {
        const float* lg = logits + lane * 16;
        float m = lg[0];
#pragma unroll
        for (int j = 1; j < 12; ++j) m = fmaxf(m, lg[j]);
        float sum = 0.f;
#pragma unroll
        for (int j = 0; j < 12; ++j) sum += __expf(lg[j] - m);
        float ls = __logf(sum) + m;
        float* op = out + ((long long)tile * 16 + lane) * 12;
#pragma unroll
        for (int j = 0; j < 12; ++j) op[j] = lg[j] - ls;
    }
}

extern "C" void kernel_launch(void* const* d_in, const int* in_sizes, int n_in,
                              void* d_out, int out_size, void* d_ws, size_t ws_size,
                              hipStream_t stream)
{
    const float* sig  = (const float*)d_in[0];
    const float* c1w  = (const float*)d_in[1];
    const float* c1b  = (const float*)d_in[2];
    const float* c2w  = (const float*)d_in[3];
    const float* c2b  = (const float*)d_in[4];
    const float* fc1w = (const float*)d_in[5];
    const float* fc1b = (const float*)d_in[6];
    const float* fc2w = (const float*)d_in[7];
    const float* fc2b = (const float*)d_in[8];
    const float* pw   = (const float*)d_in[9];
    const float* pb   = (const float*)d_in[10];
    _Float16* ws = (_Float16*)d_ws;

    const int total = F1_HALF + F2_HALF + HD_HALF;   // 115712 halves = 226 KB
    prep_weights<<<(total + 255) / 256, 256, 0, stream>>>(fc1w, fc2w, pw, ws);

    const int nblocks = B_TOTAL / 128;               // 1024 blocks, 128 samples each
    const size_t shmem = HDR_LDS + 8 * (size_t)WAVE_LDS;  // 279,552 B (< 320 KB WGP LDS)
    convnet_main<<<nblocks, 256, shmem, stream>>>(sig, c1w, c1b, c2w, c2b,
                                                  fc1b, fc2b, pb, ws, (float*)d_out);
}
